// GConvN_61512521613335
// MI455X (gfx1250) — matrix operand order repaired and checked by reading, not compiled, not run on hardware
//
#include <hip/hip_runtime.h>
#include <hip/hip_bf16.h>

typedef __attribute__((ext_vector_type(16))) _Float16 v16h;
typedef __attribute__((ext_vector_type(8)))  float    v8f;

#define NNODES 150000
#define NEDGES 1200000

static __device__ __forceinline__ void atomicAddF32(float* p, float v) {
    // relaxed, device scope -> native global_atomic_add_f32 (no CAS loop)
    (void)__hip_atomic_fetch_add(p, v, __ATOMIC_RELAXED, __HIP_MEMORY_SCOPE_AGENT);
}

// ---------------------------------------------------------------- degree ----
__global__ void deg_kernel(const int* __restrict__ ei, float* __restrict__ deg, int E) {
    int e = blockIdx.x * blockDim.x + threadIdx.x;
    if (e >= E) return;
    int dst = ei[E + e];
    atomicAddF32(&deg[dst], 1.0f);
}

__global__ void dinv_kernel(const float* __restrict__ deg, float* __restrict__ dinv, int n) {
    int i = blockIdx.x * blockDim.x + threadIdx.x;
    if (i >= n) return;
    dinv[i] = rsqrtf(deg[i] + 1.0f);   // deg includes self-loop (+1)
}

// ------------------------------------------------- GEMM1: xw = X @ W1 -------
// Block = 128 threads (4 waves). Block m-tile = 16 rows; wave w -> cols [16w,16w+16).
// K = 64 -> two v_wmma_f32_16x16x32_f16 per wave, f32 accumulate.
__global__ void gemm1_wmma(const float* __restrict__ X,
                           const float* __restrict__ W,     // [64][64] row-major (k, n)
                           float* __restrict__ xw) {
    __shared__ _Float16 Wlds[64 * 64];                      // 8 KB f16 copy of W1
    const int tid = threadIdx.x;
    for (int i = tid; i < 64 * 64; i += 128)
        Wlds[i] = (_Float16)W[i];
    __syncthreads();

    const int wave = tid >> 5;
    const int lane = tid & 31;
    const int col  = lane & 15;       // N within tile / M-row for A
    const int half = lane >> 4;       // lane group 0..15 vs 16..31
    const long long m0 = (long long)blockIdx.x * 16;
    const int n0 = wave * 16;

    v8f c = {};
#pragma unroll
    for (int kb = 0; kb < 2; ++kb) {
        const int k0 = kb * 32;
        // A fragment (16-bit A 16x32 layout): row = lane&15;
        // elements 0..7 -> K = k0 + 8*half + i ; elements 8..15 -> K = k0 + 8*half + 16 + i
        v16h a;
        const float* xr = X + (m0 + col) * 64 + k0 + half * 8;
#pragma unroll
        for (int i = 0; i < 8; ++i) a[i]     = (_Float16)xr[i];
#pragma unroll
        for (int i = 0; i < 8; ++i) a[8 + i] = (_Float16)xr[16 + i];
        // B fragment (32x16, K x N): elements 0..7 -> K = k0 + 8*half + i ; elements 8..15 -> K = k0 + 8*half + 16 + i
        v16h b;
#pragma unroll
        for (int i = 0; i < 16; ++i)
            b[i] = Wlds[(k0 + half * 8 + (i & 7) + ((i >> 3) << 4)) * 64 + (n0 + col)];
        c = __builtin_amdgcn_wmma_f32_16x16x32_f16(false, a, false, b, (short)0, c, false, false);
    }
    // C layout: VGPR v -> row M = v + 8*half, col = n0 + (lane&15)
#pragma unroll
    for (int v = 0; v < 8; ++v)
        xw[(m0 + v + 8 * half) * 64 + n0 + col] = c[v];
}

// --------------------------------------- edge scatter: agg[dst] += xw[src]*norm
// 16 lanes per edge, float4 gathers, 4 fp32 atomics per lane.
__global__ void edge_agg64(const int* __restrict__ ei, const float* __restrict__ xw,
                           const float* __restrict__ dinv, float* __restrict__ agg, int E) {
    int t = blockIdx.x * blockDim.x + threadIdx.x;      // E*16 threads
    int e = t >> 4, c = t & 15;
    if (e >= E) return;
    int s = ei[e], d = ei[E + e];
    float nrm = dinv[s] * dinv[d];
    float4 v = *(const float4*)(xw + (long long)s * 64 + c * 4);
    float* ap = agg + (long long)d * 64 + c * 4;
    atomicAddF32(ap + 0, v.x * nrm);
    atomicAddF32(ap + 1, v.y * nrm);
    atomicAddF32(ap + 2, v.z * nrm);
    atomicAddF32(ap + 3, v.w * nrm);
}

// ------------------- h1 = relu(agg + xw*dinv^2 + b1), in place over agg -----
__global__ void finalize1(float* __restrict__ agg, const float* __restrict__ xw,
                          const float* __restrict__ dinv, const float* __restrict__ b1, int n16) {
    int t = blockIdx.x * blockDim.x + threadIdx.x;      // N*16 float4 chunks
    if (t >= n16) return;
    int node = t >> 4, c = t & 15;
    float s = dinv[node]; s *= s;
    float4 a = ((const float4*)agg)[t];
    float4 x = ((const float4*)xw)[t];
    float4 bb = ((const float4*)b1)[c];
    a.x = fmaxf(fmaf(x.x, s, a.x) + bb.x, 0.0f);
    a.y = fmaxf(fmaf(x.y, s, a.y) + bb.y, 0.0f);
    a.z = fmaxf(fmaf(x.z, s, a.z) + bb.z, 0.0f);
    a.w = fmaxf(fmaf(x.w, s, a.w) + bb.w, 0.0f);
    ((float4*)agg)[t] = a;
}

// ------------------------------ layer 2 linear: z[n] = dot(h1[n,:], W2[:,0]) -
__global__ void lin2_kernel(const float* __restrict__ h1, const float* __restrict__ W2,
                            float* __restrict__ z, int n) {
    int i = blockIdx.x * blockDim.x + threadIdx.x;
    if (i >= n) return;
    const float4* h = (const float4*)(h1 + (long long)i * 64);
    const float4* w = (const float4*)W2;
    float acc = 0.0f;
#pragma unroll
    for (int k = 0; k < 16; ++k) {
        float4 a = h[k], b = w[k];
        acc += a.x * b.x + a.y * b.y + a.z * b.z + a.w * b.w;
    }
    z[i] = acc;
}

// ------------------------------- scalar edge scatter: agg2[dst] += z[src]*norm
__global__ void edge_agg1(const int* __restrict__ ei, const float* __restrict__ z,
                          const float* __restrict__ dinv, float* __restrict__ agg2, int E) {
    int e = blockIdx.x * blockDim.x + threadIdx.x;
    if (e >= E) return;
    int s = ei[e], d = ei[E + e];
    atomicAddF32(&agg2[d], z[s] * dinv[s] * dinv[d]);
}

// ---- out[i*12+j] = relu(agg2[n] + z[n]*dinv^2 + b2),  n = i*15 + 3 + j ------
__global__ void finalize2_slice(const float* __restrict__ agg2, const float* __restrict__ z,
                                const float* __restrict__ dinv, const float* __restrict__ b2,
                                float* __restrict__ out, int nout) {
    int t = blockIdx.x * blockDim.x + threadIdx.x;
    if (t >= nout) return;
    int i = t / 12, j = t - i * 12;
    int node = i * 15 + 3 + j;
    float s = dinv[node]; s *= s;
    float v = agg2[node] + z[node] * s + b2[0];
    out[t] = fmaxf(v, 0.0f);
}

extern "C" void kernel_launch(void* const* d_in, const int* in_sizes, int n_in,
                              void* d_out, int out_size, void* d_ws, size_t ws_size,
                              hipStream_t stream) {
    const float* obs = (const float*)d_in[0];
    const int*   ei  = (const int*)d_in[1];   // [2][E]: row0 = src, row1 = dst
    const float* W1  = (const float*)d_in[2];
    const float* b1  = (const float*)d_in[3];
    const float* W2  = (const float*)d_in[4];
    const float* b2  = (const float*)d_in[5];
    float* out = (float*)d_out;
    float* ws  = (float*)d_ws;

    const int Nn = NNODES, Ee = NEDGES;
    float* degz = ws;                              // N   (degree, later reused for z)
    float* dinv = ws + (size_t)Nn;                 // N
    float* xw   = ws + (size_t)2 * Nn;             // N*64
    float* agg1 = ws + (size_t)66 * Nn;            // N*64 (becomes h1 in place)
    float* agg2 = ws + (size_t)130 * Nn;           // N

    (void)hipMemsetAsync(degz, 0, (size_t)Nn * sizeof(float), stream);
    (void)hipMemsetAsync(agg1, 0, (size_t)Nn * 64 * sizeof(float), stream);
    (void)hipMemsetAsync(agg2, 0, (size_t)Nn * sizeof(float), stream);

    deg_kernel <<<(Ee + 255) / 256, 256, 0, stream>>>(ei, degz, Ee);
    dinv_kernel<<<(Nn + 255) / 256, 256, 0, stream>>>(degz, dinv, Nn);

    gemm1_wmma<<<Nn / 16, 128, 0, stream>>>(obs, W1, xw);

    edge_agg64<<<(Ee * 16 + 255) / 256, 256, 0, stream>>>(ei, xw, dinv, agg1, Ee);
    finalize1 <<<(Nn * 16 + 255) / 256, 256, 0, stream>>>(agg1, xw, dinv, b1, Nn * 16);

    lin2_kernel<<<(Nn + 255) / 256, 256, 0, stream>>>(agg1, W2, degz, Nn);  // z -> degz
    edge_agg1  <<<(Ee + 255) / 256, 256, 0, stream>>>(ei, degz, dinv, agg2, Ee);

    const int nout = (Nn / 15) * 12;  // 120000
    finalize2_slice<<<(nout + 255) / 256, 256, 0, stream>>>(agg2, degz, dinv, b2, out, nout);
}
